// AttentionHead_58445914964473
// MI455X (gfx1250) — matrix operand order repaired and checked by reading, not compiled, not run on hardware
//
#include <hip/hip_runtime.h>
#include <math.h>

// Problem constants (match reference)
#define Bn 4
#define Sn 4096
#define En 1024
#define Dn 64

typedef __bf16 v16bf __attribute__((ext_vector_type(16)));
typedef float  v8f   __attribute__((ext_vector_type(8)));

// ---- WMMA 16x16x32 bf16 helper ---------------------------------------------
__device__ __forceinline__ v8f wmma_bf16(v16bf a, v16bf b, v8f c) {
    return __builtin_amdgcn_wmma_f32_16x16x32_bf16(
        /*neg_a=*/false, a, /*neg_b=*/false, b,
        /*c_mod=*/(short)0, c, /*reuse_a=*/false, /*reuse_b=*/false);
}

// A-matrix 16-bit 16x32 layout used here (contiguous k, not the ISA table order):
//  lane L: M = L%16, half = L/16
//  VGPR v (v<8):  K = 2v + {0,1} + 16*half  (contiguous, matches B fragments)
__device__ __forceinline__ int amapK(int v, int half) {
    int base = 2 * v;
    return base + 16 * half;
}

// 16-lane row max (rows of a C tile live across 16 consecutive lanes;
// xor masks 1..8 never flip lane bit 4, so they stay within each half-wave).
__device__ __forceinline__ float redmax16(float x) {
#pragma unroll
    for (int m = 8; m >= 1; m >>= 1) x = fmaxf(x, __shfl_xor(x, m, 32));
    return x;
}

// LDS offset of a flat shared-memory pointer (ISA 10.2: flat LDS addresses
// carry the workgroup-relative LDS byte offset in addr[31:0]).
__device__ __forceinline__ unsigned lds_off(const void* p) {
    return (unsigned)(uintptr_t)p;
}

// ---- Phase 0: swizzle weights [E,D] f32 -> B-fragment bf16 ------------------
// B-matrix 32x16 layout: lane L: n = L%16, khalf = L/16; element e: k = 16*khalf + e.
// Fragment array: [3][E/32 kb][4 dtile][32 lane][16 e]
__global__ __launch_bounds__(256) void swizzle_weights(
    const float* __restrict__ wq, const float* __restrict__ wk,
    const float* __restrict__ wv, __bf16* __restrict__ wfrag) {
    const int per = (En / 32) * 4 * 32 * 16;  // 65536
    int idx = blockIdx.x * blockDim.x + threadIdx.x;
    if (idx >= 3 * per) return;
    int w = idx / per;
    int r = idx % per;
    int e = r & 15;
    int lane = (r >> 4) & 31;
    int t = (r >> 9) & 3;
    int kbb = (r >> 11) & 31;
    const float* W = (w == 0) ? wq : (w == 1) ? wk : wv;
    int n = lane & 15, khalf = lane >> 4;
    int k = khalf * 16 + e;
    float val = W[(size_t)(kbb * 32 + k) * Dn + t * 16 + n];
    wfrag[(size_t)w * per + ((size_t)(kbb * 4 + t) * 32 + lane) * 16 + e] = (__bf16)val;
}

// ---- Phase 1: projections  out = in[16384,1024] @ w[1024,64] ----------------
// Activation tiles stream global->LDS via async copies (ASYNCcnt), double
// buffered so stage kb+1 is in flight while WMMAs consume stage kb.
// mode 0: qh  -> A-fragment layout      [B*S/16 tile][2 kblk][32 lane][16]
// mode 1: khT -> B-fragment (K=d)       [B*S/16 tile][2 kblk][32 lane][16]
// mode 2: vh  -> B-fragment (K=key)     [B*S/32 blk ][4 dtile][32 lane][16]
__global__ __launch_bounds__(256) void project(
    const float* __restrict__ q, const float* __restrict__ k,
    const float* __restrict__ v, const __bf16* __restrict__ wfrag,
    __bf16* __restrict__ qh, __bf16* __restrict__ kh, __bf16* __restrict__ vh) {
    const int mode = blockIdx.y;
    const float* in = (mode == 0) ? q : (mode == 1) ? k : v;
    const __bf16* wf = wfrag + (size_t)mode * ((En / 32) * 4 * 512);

    __shared__ float As[2][128 * 32];     // 2 x 16 KB double-buffered staging
    __shared__ __bf16 Cs[8][16 * 64];     // 16 KB per-wave C tiles (row-major)

    const int tid = threadIdx.x, wid = tid >> 5, lane = tid & 31;
    const int half = lane >> 4, Ml = lane & 15, n = lane & 15;
    const int row0 = blockIdx.x * 128;    // 8 row-tiles of 16 per workgroup
    const int NK = En / 32;               // 32 K-steps

    // async global->LDS copy of one 128x32 f32 tile (4x b128 per thread)
    auto issue_stage = [&](int kb, int buf) {
#pragma unroll
        for (int i = 0; i < 4; ++i) {
            int c = tid + 256 * i;            // float4 chunk id, 8 per row
            int rr = c >> 3;
            int cc = (c & 7) << 2;
            const float* g = in + (size_t)(row0 + rr) * En + kb * 32 + cc;
            unsigned lo = lds_off(&As[buf][rr * 32 + cc]);
            asm volatile("global_load_async_to_lds_b128 %0, %1, off"
                         :: "v"(lo), "v"(g) : "memory");
        }
    };

    issue_stage(0, 0);

    v8f acc0 = {}, acc1 = {}, acc2 = {}, acc3 = {};

    for (int kb = 0; kb < NK; ++kb) {
        const int cur = kb & 1;
        if (kb + 1 < NK) {
            issue_stage(kb + 1, cur ^ 1);
            // 8 asyncs outstanding; <=4 left means stage kb fully landed in LDS
            asm volatile("s_wait_asynccnt 0x4" ::: "memory");
        } else {
            asm volatile("s_wait_asynccnt 0x0" ::: "memory");
        }
        __syncthreads();                  // all waves' stage-kb data visible

        // build A fragment for this wave's 16 rows (f32 -> bf16)
        v16bf a;
#pragma unroll
        for (int vv = 0; vv < 8; ++vv) {
            int K = amapK(vv, half);
            a[2 * vv]     = (__bf16)As[cur][(wid * 16 + Ml) * 32 + K];
            a[2 * vv + 1] = (__bf16)As[cur][(wid * 16 + Ml) * 32 + K + 1];
        }
        // pre-swizzled weight B fragments: 32 B/lane contiguous
        v16bf b0 = *(const v16bf*)(wf + ((size_t)(kb * 4 + 0) * 32 + lane) * 16);
        v16bf b1 = *(const v16bf*)(wf + ((size_t)(kb * 4 + 1) * 32 + lane) * 16);
        v16bf b2 = *(const v16bf*)(wf + ((size_t)(kb * 4 + 2) * 32 + lane) * 16);
        v16bf b3 = *(const v16bf*)(wf + ((size_t)(kb * 4 + 3) * 32 + lane) * 16);
        acc0 = wmma_bf16(a, b0, acc0);
        acc1 = wmma_bf16(a, b1, acc1);
        acc2 = wmma_bf16(a, b2, acc2);
        acc3 = wmma_bf16(a, b3, acc3);

        __syncthreads();                  // reads done; buffer may be rewritten
    }

    // C layout: lane L, VGPR r -> (M = r + 8*(L/16), N = L%16)
#pragma unroll
    for (int r = 0; r < 8; ++r) {
        int M = r + 8 * half;
        Cs[wid][M * 64 + 0 * 16 + n] = (__bf16)acc0[r];
        Cs[wid][M * 64 + 1 * 16 + n] = (__bf16)acc1[r];
        Cs[wid][M * 64 + 2 * 16 + n] = (__bf16)acc2[r];
        Cs[wid][M * 64 + 3 * 16 + n] = (__bf16)acc3[r];
    }
    __syncthreads();

    if (mode == 0) {
        // Qh as A fragments (K dim = d, 2 blocks of 32)
        int tile = row0 / 16 + wid;
#pragma unroll
        for (int kblk = 0; kblk < 2; ++kblk) {
            v16bf o;
#pragma unroll
            for (int vv = 0; vv < 8; ++vv) {
                int K = amapK(vv, half);
                o[2 * vv]     = Cs[wid][Ml * 64 + kblk * 32 + K];
                o[2 * vv + 1] = Cs[wid][Ml * 64 + kblk * 32 + K + 1];
            }
            *(v16bf*)(qh + ((size_t)(tile * 2 + kblk) * 32 + lane) * 16) = o;
        }
    } else if (mode == 1) {
        // Kh^T as B fragments (K dim = d, N = key)
        int tile = row0 / 16 + wid;
#pragma unroll
        for (int kblk = 0; kblk < 2; ++kblk) {
            v16bf o;
#pragma unroll
            for (int e = 0; e < 16; ++e) {
                int d = kblk * 32 + half * 16 + e;
                o[e] = Cs[wid][n * 64 + d];   // key-local row n, column d
            }
            *(v16bf*)(kh + ((size_t)(tile * 2 + kblk) * 32 + lane) * 16) = o;
        }
    } else {
        // Vh as B fragments (K dim = key block of 32, N = d tile of 16)
        int j = wid >> 1;                 // 32-key block within this WG
        int blk = row0 / 32 + j;
        int t0 = (wid & 1) * 2;           // each wave of the pair writes 2 dtiles
#pragma unroll
        for (int tt = 0; tt < 2; ++tt) {
            int t = t0 + tt;
            v16bf o;
#pragma unroll
            for (int e = 0; e < 16; ++e) {
                // element (k = half*16 + e, n): key row e of wave (2j + half)
                o[e] = Cs[2 * j + half][e * 64 + t * 16 + n];
            }
            *(v16bf*)(vh + ((size_t)(blk * 4 + t) * 32 + lane) * 16) = o;
        }
    }
}

// ---- Phase 2: fused causal flash attention ---------------------------------
// 1 wave = one 16-row query tile; O = 16x64 in 4 f32 accumulators.
// Softmax denominator computed as a 5th WMMA (P @ ones), rescaled like O.
__global__ __launch_bounds__(256) void attention(
    const __bf16* __restrict__ qh, const __bf16* __restrict__ kh,
    const __bf16* __restrict__ vh, float* __restrict__ out) {
    __shared__ __bf16 Ps[8][16 * 32];     // wave-private P staging (8 KB)

    const int tid = threadIdx.x, wid = tid >> 5, lane = tid & 31;
    const int half = lane >> 4, n = lane & 15;
    const int b = blockIdx.y;
    const int qt = blockIdx.x * 8 + wid;  // query tile within batch
    const int q0 = qt * 16;

    const __bf16* qhb = qh + (size_t)b * (Sn / 16) * 2 * 512;
    const __bf16* khb = kh + (size_t)b * (Sn / 16) * 2 * 512;
    const __bf16* vhb = vh + (size_t)b * (Sn / 32) * 4 * 512;

    const v16bf qa0 = *(const v16bf*)(qhb + ((size_t)(qt * 2 + 0) * 32 + lane) * 16);
    const v16bf qa1 = *(const v16bf*)(qhb + ((size_t)(qt * 2 + 1) * 32 + lane) * 16);

    v16bf ones;
#pragma unroll
    for (int e = 0; e < 16; ++e) ones[e] = (__bf16)1.0f;

    v8f O0 = {}, O1 = {}, O2 = {}, O3 = {}, Lacc = {};
    float m[8];
#pragma unroll
    for (int r = 0; r < 8; ++r) m[r] = -__builtin_inff();

    const int nkb = (q0 + 15) / 32 + 1;   // causal: only key blocks <= q0+15
    for (int kb = 0; kb < nkb; ++kb) {
        // prefetch next key block's fragments while we compute
        if (kb + 1 < nkb) {
            __builtin_prefetch(khb + ((size_t)((kb + 1) * 2) * 32 + lane) * 16, 0, 0);
            __builtin_prefetch(vhb + ((size_t)(kb + 1) * 4 * 512) + lane * 16, 0, 0);
        }
        float st[2][8];
        float bm[8];
#pragma unroll
        for (int r = 0; r < 8; ++r) bm[r] = -__builtin_inff();

#pragma unroll
        for (int sub = 0; sub < 2; ++sub) {
            int kt = kb * 2 + sub;
            v16bf k0 = *(const v16bf*)(khb + ((size_t)(kt * 2 + 0) * 32 + lane) * 16);
            v16bf k1 = *(const v16bf*)(khb + ((size_t)(kt * 2 + 1) * 32 + lane) * 16);
            v8f s = {};
            s = wmma_bf16(qa0, k0, s);
            s = wmma_bf16(qa1, k1, s);
            int key = kt * 16 + n;
#pragma unroll
            for (int r = 0; r < 8; ++r) {
                int qrow = q0 + r + 8 * half;
                float x = (key <= qrow) ? s[r] * 0.125f : -__builtin_inff();
                st[sub][r] = x;
                bm[r] = fmaxf(bm[r], x);
            }
        }
        // online softmax rescale (per-row stats replicated across 16 lanes)
        float alpha[8];
#pragma unroll
        for (int r = 0; r < 8; ++r) {
            bm[r] = redmax16(bm[r]);
            float mn = fmaxf(m[r], bm[r]);
            alpha[r] = __expf(m[r] - mn);
            m[r] = mn;
        }
        float p0[8], p1[8];
#pragma unroll
        for (int r = 0; r < 8; ++r) {
            p0[r] = __expf(st[0][r] - m[r]);
            p1[r] = __expf(st[1][r] - m[r]);
            O0[r] *= alpha[r]; O1[r] *= alpha[r];
            O2[r] *= alpha[r]; O3[r] *= alpha[r];
            Lacc[r] *= alpha[r];
        }
        // route P through wave-private LDS into A-fragment layout
        // (same-wave LDS ops are kept in order by hardware; no barrier needed)
#pragma unroll
        for (int r = 0; r < 8; ++r) {
            int M = r + 8 * half;
            Ps[wid][M * 32 + 0 * 16 + n] = (__bf16)p0[r];
            Ps[wid][M * 32 + 1 * 16 + n] = (__bf16)p1[r];
        }
        v16bf pa;
#pragma unroll
        for (int vv = 0; vv < 8; ++vv) {
            int K = amapK(vv, half);
            pa[2 * vv]     = Ps[wid][(lane & 15) * 32 + K];
            pa[2 * vv + 1] = Ps[wid][(lane & 15) * 32 + K + 1];
        }
        // O += P @ V ; Lacc += P @ ones (row sums on the matrix pipe)
        const __bf16* vb = vhb + (size_t)kb * 4 * 512;
        v16bf v0 = *(const v16bf*)(vb + ((size_t)(0) * 32 + lane) * 16);
        v16bf v1 = *(const v16bf*)(vb + ((size_t)(1) * 32 + lane) * 16);
        v16bf v2 = *(const v16bf*)(vb + ((size_t)(2) * 32 + lane) * 16);
        v16bf v3 = *(const v16bf*)(vb + ((size_t)(3) * 32 + lane) * 16);
        O0 = wmma_bf16(pa, v0, O0);
        O1 = wmma_bf16(pa, v1, O1);
        O2 = wmma_bf16(pa, v2, O2);
        O3 = wmma_bf16(pa, v3, O3);
        Lacc = wmma_bf16(pa, ones, Lacc);
    }

    // epilogue: normalize and store f32 [B,S,D]
#pragma unroll
    for (int r = 0; r < 8; ++r) {
        float inv = 1.0f / Lacc[r];
        int row = q0 + r + 8 * half;
        float* orow = out + ((size_t)b * Sn + row) * Dn;
        orow[0 * 16 + n] = O0[r] * inv;
        orow[1 * 16 + n] = O1[r] * inv;
        orow[2 * 16 + n] = O2[r] * inv;
        orow[3 * 16 + n] = O3[r] * inv;
    }
}

// ---- Host launcher ----------------------------------------------------------
extern "C" void kernel_launch(void* const* d_in, const int* in_sizes, int n_in,
                              void* d_out, int out_size, void* d_ws, size_t ws_size,
                              hipStream_t stream) {
    (void)in_sizes; (void)n_in; (void)out_size; (void)ws_size;
    const float* q  = (const float*)d_in[0];
    const float* k  = (const float*)d_in[1];
    const float* v  = (const float*)d_in[2];
    // d_in[3] = causal mask (tril) — equivalent to the arithmetic causal test; unused
    const float* wq = (const float*)d_in[4];
    const float* wk = (const float*)d_in[5];
    const float* wv = (const float*)d_in[6];
    float* out = (float*)d_out;

    char* ws = (char*)d_ws;
    __bf16* wfrag = (__bf16*)ws;                                 // 384 KB
    __bf16* qh    = (__bf16*)(ws + 393216);                      // 2 MB
    __bf16* kh    = (__bf16*)(ws + 393216 + 2097152);            // 2 MB
    __bf16* vh    = (__bf16*)(ws + 393216 + 2 * 2097152);        // 2 MB

    swizzle_weights<<<dim3(768), dim3(256), 0, stream>>>(wq, wk, wv, wfrag);
    project<<<dim3((Bn * Sn) / 128, 3), dim3(256), 0, stream>>>(q, k, v, wfrag, qh, kh, vh);
    attention<<<dim3(Sn / 128, Bn), dim3(256), 0, stream>>>(qh, kh, vh, out);
}
